// ExtendedMixtureRNN_40656160424382
// MI455X (gfx1250) — matrix operand order repaired and checked by reading, not compiled, not run on hardware
//
#include <hip/hip_runtime.h>
#include <stdint.h>

#define Bn 256
#define Tn 512
#define Dn 1024
#define Vn 32000
#define KT (Dn / 32)       // 32 k-tiles of depth 32
#define EPSF 1e-5f

typedef __attribute__((ext_vector_type(16))) __bf16 v16bf;
typedef __attribute__((ext_vector_type(8)))  float  v8f;
typedef __attribute__((ext_vector_type(4)))  unsigned int v4u;
typedef __attribute__((ext_vector_type(8)))  int    v8i;
typedef __attribute__((ext_vector_type(4)))  int    v4i;

// ---- fp32 -> bf16 (round to nearest even), branch-free, no libcalls ----
__device__ __forceinline__ unsigned short bf16_bits(float f) {
  unsigned u = __builtin_bit_cast(unsigned, f);
  unsigned r = u + 0x7FFFu + ((u >> 16) & 1u);
  return (unsigned short)(r >> 16);
}
__device__ __forceinline__ __bf16 to_bf16(float f) {
  unsigned short h = bf16_bits(f);
  return __builtin_bit_cast(__bf16, h);
}

// ---------------- zero init of state z ----------------
__global__ void emr_zero_kernel(float* __restrict__ z, int n) {
  int idx = blockIdx.x * 256 + threadIdx.x;
  if (idx < n) z[idx] = 0.f;
}

// ---------------- pack W1 = W_dt[0:D,:] into WMMA B-fragment layout (bf16) ----------------
// B tile (32xK,16xN): lane = ((kk>>4)<<4) | (n&15); half j = kk&15  (kk = K within tile)
__global__ void emr_packW1_kernel(const float* __restrict__ Wdt, unsigned short* __restrict__ w1pack) {
  int idx = blockIdx.x * 256 + threadIdx.x;
  if (idx >= Dn * Dn) return;
  int k = idx >> 10, n = idx & (Dn - 1);
  int tile_n = n >> 4, tile_k = k >> 5, kk = k & 31;
  int lane = ((kk >> 4) << 4) | (n & 15);
  int j = kk & 15;
  w1pack[(((tile_n * KT + tile_k) * 32) + lane) * 16 + j] = bf16_bits(Wdt[idx]);
}

// ---------------- kproj = hidden @ W_dt[D:2D,:] + b_dt  (one-time, loop invariant) ----------------
__global__ void emr_kproj_kernel(const float* __restrict__ hs, const float* __restrict__ Wdt,
                                 const float* __restrict__ bdt, float* __restrict__ kproj) {
  int idx = blockIdx.x * 256 + threadIdx.x;
  if (idx >= Bn * Dn) return;
  int b = idx >> 10, f = idx & (Dn - 1);
  float acc = bdt[f];
  const float* kr = hs + b * Dn;
  for (int d = 0; d < Dn; ++d)
    acc += kr[d] * Wdt[(Dn + d) * Dn + f];
  kproj[idx] = acc;
}

// ---------------- u = z + emb[seq[:,t]] ; also write u in WMMA A-fragment layout (bf16) ----------------
// A tile (16xM,32xK): lane = ((kk>>4)<<4)|(m&15); half j = kk&15
__global__ void emr_u_kernel(const float* __restrict__ z, const float* __restrict__ emb,
                             const int* __restrict__ seq, int t,
                             float* __restrict__ u, unsigned short* __restrict__ apack) {
  int idx = blockIdx.x * 256 + threadIdx.x;
  if (idx >= Bn * Dn) return;
  int b = idx >> 10, d = idx & (Dn - 1);
  int tok = seq[b * Tn + t];
  float v = z[idx] + emb[(size_t)tok * Dn + d];
  u[idx] = v;
  int mt = b >> 4, kt = d >> 5, kk = d & 31;
  int lane = ((kk >> 4) << 4) | (b & 15);
  int j = kk & 15;
  apack[(((mt * KT + kt) * 32) + lane) * 16 + j] = bf16_bits(v);
}

// ---------------- per-step GEMM: pre = u @ W1 + kproj + u  (WMMA bf16, f32 accumulate) ----------------
// 32 blocks x 8 waves; all 8 waves of a block share one 16-row A strip (mt = blockIdx/2),
// which in packed layout is one CONTIGUOUS 32KB region -> staged into LDS by the
// Tensor Data Mover (1-D D#: data_size=8B, tile_dim0=tensor_dim0=4096), then each
// wave reads A fragments from LDS (ds_load_b128) while streaming B from L2.
__global__ void emr_step_gemm_kernel(const unsigned short* __restrict__ Apack,
                                     const v16bf* __restrict__ Bpack,
                                     const float* __restrict__ kproj, const float* __restrict__ u,
                                     float* __restrict__ pre) {
  __shared__ __align__(64) unsigned short lds_a[KT * 32 * 16];   // 32 KB A strip
  int lane = threadIdx.x & 31;
  int wave = threadIdx.x >> 5;
  int w = blockIdx.x * 8 + wave;
  int mt = w >> 4;          // uniform across the block (8 waves per block, 16 per mt)
  int ng = w & 15;          // 0..15 : 64-col group

  if (wave == 0) {
    unsigned long long gaddr =
        (unsigned long long)(uintptr_t)(Apack + (size_t)mt * KT * 32 * 16);
    unsigned int laddr = (unsigned int)(uintptr_t)lds_a;
    // D# group0: count=1 | lds_addr | global_addr[56:0] | type=2
    v4u g0 = { 1u, laddr, (unsigned int)gaddr,
               (unsigned int)((gaddr >> 32) & 0x1FFFFFFu) | (2u << 30) };
    // D# group1: data_size=3 (8B); tensor_dim0 = tile_dim0 = 4096 elements (32KB);
    // 1-D tile (tile_dim1/2 = 0), stride0 = 4096, no multicast/pad/iterate.
    v8i g1 = { (int)(3u << 16),
               (int)((4096u & 0xFFFFu) << 16),   // tensor_dim0[15:0] in [31:16]
               0,                                 // tensor_dim0[31:16]=0, tensor_dim1[15:0]=0
               (int)(4096u << 16),                // tile_dim0 in [31:16]
               0,                                 // tile_dim1=0, tile_dim2=0
               4096,                              // tensor_dim0_stride[31:0]
               0, 0 };
    v4i g2 = { 0, 0, 0, 0 };
    v4i g3 = { 0, 0, 0, 0 };
    v8i g4 = { 0, 0, 0, 0, 0, 0, 0, 0 };
    __builtin_amdgcn_tensor_load_to_lds(g0, g1, g2, g3, g4, 0);
    __builtin_amdgcn_s_wait_tensorcnt(0);
  }
  __syncthreads();

  const v16bf* lA = (const v16bf*)lds_a;
  v8f c[4] = {};
  for (int kt = 0; kt < KT; ++kt) {
    v16bf bf[4];
#pragma unroll
    for (int j = 0; j < 4; ++j)
      bf[j] = Bpack[((ng * 4 + j) * KT + kt) * 32 + lane];
    v16bf a = lA[kt * 32 + lane];
#pragma unroll
    for (int j = 0; j < 4; ++j)
      c[j] = __builtin_amdgcn_wmma_f32_16x16x32_bf16(false, a, false, bf[j], (short)0, c[j], false, false);
  }
  // C/D layout: VGPR i, lanes 0-15 -> M=i ; lanes 16-31 -> M=8+i ; N = lane&15
  int rbase = mt * 16 + ((lane >> 4) << 3);
  int colb = lane & 15;
#pragma unroll
  for (int j = 0; j < 4; ++j) {
    int col = ng * 64 + j * 16 + colb;
#pragma unroll
    for (int i = 0; i < 8; ++i) {
      int idx = (rbase + i) * Dn + col;
      pre[idx] = c[j][i] + kproj[idx] + u[idx];
    }
  }
}

// ---------------- LayerNorm over D per row ----------------
__global__ void emr_ln_kernel(const float* __restrict__ pre, const float* __restrict__ gamma,
                              const float* __restrict__ beta, float* __restrict__ z) {
  int b = blockIdx.x, tid = threadIdx.x;
  __shared__ float s1[256], s2[256];
  const float* row = pre + b * Dn;
  float x[4], s = 0.f, ss = 0.f;
#pragma unroll
  for (int i = 0; i < 4; ++i) { x[i] = row[tid + i * 256]; s += x[i]; ss += x[i] * x[i]; }
  s1[tid] = s; s2[tid] = ss; __syncthreads();
  for (int off = 128; off > 0; off >>= 1) {
    if (tid < off) { s1[tid] += s1[tid + off]; s2[tid] += s2[tid + off]; }
    __syncthreads();
  }
  float m = s1[0] * (1.f / Dn);
  float var = s2[0] * (1.f / Dn) - m * m;
  float rs = rsqrtf(var + EPSF);
#pragma unroll
  for (int i = 0; i < 4; ++i) {
    int d = tid + i * 256;
    z[b * Dn + d] = (x[i] - m) * rs * gamma[d] + beta[d];
  }
}

// ---------------- write final z to d_out and pack z (bf16, A-fragment layout) ----------------
__global__ void emr_zfinal_kernel(const float* __restrict__ z, float* __restrict__ outz,
                                  unsigned short* __restrict__ apack) {
  int idx = blockIdx.x * 256 + threadIdx.x;
  if (idx >= Bn * Dn) return;
  float v = z[idx];
  outz[idx] = v;
  int b = idx >> 10, d = idx & (Dn - 1);
  int mt = b >> 4, kt = d >> 5, kk = d & 31;
  int lane = ((kk >> 4) << 4) | (b & 15);
  int j = kk & 15;
  apack[(((mt * KT + kt) * 32) + lane) * 16 + j] = bf16_bits(v);
}

// ---------------- logits = z @ W_v + b_v  (HBM-bound: stream W_v once, cvt to bf16 in-flight) ----------------
// One wave per 16-column tile of V (2000 waves, 250 blocks); wave carries all 16 M-tiles (128 acc VGPRs).
__global__ void emr_logits_kernel(const v16bf* __restrict__ Zpack, const float* __restrict__ Wv,
                                  const float* __restrict__ bv, float* __restrict__ y) {
  int lane = threadIdx.x & 31;
  int tn = blockIdx.x * 8 + (threadIdx.x >> 5);   // 0..1999
  int col = tn * 16 + (lane & 15);
  int khalf = (lane >> 4) << 4;
  v8f c[16] = {};
  for (int kt = 0; kt < KT; ++kt) {
    int kbase = kt * 32 + khalf;
    if (kt + 1 < KT)
      __builtin_prefetch(&Wv[(size_t)(kbase + 32) * Vn + col], 0, 1);   // global_prefetch_b8
    v16bf bf;
#pragma unroll
    for (int q = 0; q < 16; ++q)
      bf[q] = to_bf16(Wv[(size_t)(kbase + q) * Vn + col]);
#pragma unroll
    for (int mt = 0; mt < 16; ++mt) {
      v16bf a = Zpack[(mt * KT + kt) * 32 + lane];
      c[mt] = __builtin_amdgcn_wmma_f32_16x16x32_bf16(false, a, false, bf, (short)0, c[mt], false, false);
    }
  }
  float bias = bv[col];
  int rb = (lane >> 4) << 3;
#pragma unroll
  for (int mt = 0; mt < 16; ++mt)
#pragma unroll
    for (int i = 0; i < 8; ++i)
      y[(size_t)(mt * 16 + rb + i) * Vn + col] = c[mt][i] + bias;
}

// ---------------- in-place log_softmax over V per row ----------------
__global__ void emr_logsoftmax_kernel(float* __restrict__ y) {
  int b = blockIdx.x, tid = threadIdx.x;
  __shared__ float red[256];
  float* row = y + (size_t)b * Vn;
  float m = -3.402823466e38f;
  for (int i = tid; i < Vn; i += 256) m = fmaxf(m, row[i]);
  red[tid] = m; __syncthreads();
  for (int off = 128; off > 0; off >>= 1) {
    if (tid < off) red[tid] = fmaxf(red[tid], red[tid + off]);
    __syncthreads();
  }
  m = red[0]; __syncthreads();
  float s = 0.f;
  for (int i = tid; i < Vn; i += 256) s += expf(row[i] - m);
  red[tid] = s; __syncthreads();
  for (int off = 128; off > 0; off >>= 1) {
    if (tid < off) red[tid] += red[tid + off];
    __syncthreads();
  }
  float lse = logf(red[0]) + m;
  __syncthreads();
  for (int i = tid; i < Vn; i += 256) row[i] = row[i] - lse;
}

extern "C" void kernel_launch(void* const* d_in, const int* in_sizes, int n_in,
                              void* d_out, int out_size, void* d_ws, size_t ws_size,
                              hipStream_t stream) {
  (void)in_sizes; (void)n_in; (void)out_size; (void)ws_size;
  const float* hs    = (const float*)d_in[0];   // hidden_state [B,1,D]
  const int*   seq   = (const int*)d_in[1];     // output_sequence [B,T]
  const float* emb   = (const float*)d_in[2];   // emb_out [V,D]
  const float* Wdt   = (const float*)d_in[3];   // W_dt [2D,D]
  const float* bdt   = (const float*)d_in[4];
  const float* gamma = (const float*)d_in[5];
  const float* beta  = (const float*)d_in[6];
  const float* Wv    = (const float*)d_in[7];   // W_v [D,V]
  const float* bv    = (const float*)d_in[8];

  float* out_z = (float*)d_out;                     // [B*D]
  float* out_y = out_z + (size_t)Bn * Dn;           // [B*V]

  char* ws = (char*)d_ws;
  float* z     = (float*)(ws);                      // 1 MB
  float* u     = (float*)(ws + (1u << 20));         // 1 MB
  float* pre   = (float*)(ws + (2u << 20));         // 1 MB
  float* kproj = (float*)(ws + (3u << 20));         // 1 MB
  unsigned short* apack  = (unsigned short*)(ws + (4u << 20));                  // 512 KB
  unsigned short* w1pack = (unsigned short*)(ws + (4u << 20) + (512u << 10));   // 2 MB

  dim3 blk(256);
  emr_zero_kernel<<<(Bn * Dn) / 256, blk, 0, stream>>>(z, Bn * Dn);
  emr_packW1_kernel<<<(Dn * Dn) / 256, blk, 0, stream>>>(Wdt, w1pack);
  emr_kproj_kernel<<<(Bn * Dn) / 256, blk, 0, stream>>>(hs, Wdt, bdt, kproj);

  for (int t = 0; t < Tn; ++t) {
    emr_u_kernel<<<(Bn * Dn) / 256, blk, 0, stream>>>(z, emb, seq, t, u, apack);
    emr_step_gemm_kernel<<<32, blk, 0, stream>>>(apack, (const v16bf*)w1pack,
                                                 kproj, u, pre);
    emr_ln_kernel<<<Bn, blk, 0, stream>>>(pre, gamma, beta, z);
  }

  emr_zfinal_kernel<<<(Bn * Dn) / 256, blk, 0, stream>>>(z, out_z, apack);
  emr_logits_kernel<<<250, blk, 0, stream>>>((const v16bf*)apack, Wv, bv, out_y);
  emr_logsoftmax_kernel<<<Bn, blk, 0, stream>>>(out_y);
}
